// MHAT2_31894427140497
// MI455X (gfx1250) — matrix operand order repaired and checked by reading, not compiled, not run on hardware
//
#include <hip/hip_runtime.h>
#include <hip/hip_bf16.h>

typedef unsigned short US;
typedef __attribute__((ext_vector_type(16))) __bf16 v16bf;
typedef __attribute__((ext_vector_type(8)))  float  v8f;

union Frag { uint4 u[2]; v16bf v; };

#define BB 2048
#define NN 64
#define QS 512
#define KS 512
#define HD 128
#define NH 8
#define CC 1024   // NH*HD
#define OS 512

__device__ inline US f2bf(float f) {
  unsigned int u = __float_as_uint(f);
  unsigned int r = u + 0x7fffu + ((u >> 16) & 1u);
  return (US)(r >> 16);
}

__device__ inline v8f wmma_bf(const Frag& a, const Frag& b, v8f c) {
  return __builtin_amdgcn_wmma_f32_16x16x32_bf16(false, a.v, false, b.v,
                                                 (short)0, c, false, false);
}

// ---------------- prep: f32 -> bf16, weights pre-transposed to BT=[N][K] ---
__global__ __launch_bounds__(256) void kPrep(const float* me, const float* Wq,
                                             const float* Wk, const float* Wv,
                                             const float* Wo, US* me_bf,
                                             US* WqT, US* Wk_bf, US* WvT, US* WoT) {
  int t = blockIdx.x * 256 + threadIdx.x;
  const int n_me = BB * QS;            // 1048576
  const int n_wq = CC * QS;            // 524288  WqT[j][c]
  const int n_wk = NH * KS * HD;       // 524288  same layout, bf16
  const int n_wv = NH * HD * KS;       // 524288  WvT[h][d][c]
  const int n_wo = OS * CC;            // 524288  WoT[o][j]
  if (t < n_me) { me_bf[t] = f2bf(me[t]); return; }
  t -= n_me;
  if (t < n_wq) { int n = t >> 9, k = t & 511; int h = n >> 7, d = n & 127;
                  WqT[t] = f2bf(Wq[(h * QS + k) * HD + d]); return; }
  t -= n_wq;
  if (t < n_wk) { Wk_bf[t] = f2bf(Wk[t]); return; }
  t -= n_wk;
  if (t < n_wv) { int h = t >> 16; int d = (t >> 9) & 127; int c = t & 511;
                  WvT[t] = f2bf(Wv[(h * KS + c) * HD + d]); return; }
  t -= n_wv;
  if (t < n_wo) { int n = t >> 10, k = t & 1023;
                  WoT[t] = f2bf(Wo[k * OS + n]); return; }
}

// ---------------- kQ: q = me@Wq + bq ; p[h] = q[:,h]@Wk[h]^T ----------------
__global__ __launch_bounds__(256) void kQ(const US* me_bf, const US* WqT,
                                          const US* Wk_bf, const float* bq,
                                          US* p_ws) {
  __shared__ __align__(16) US me_s[16 * 512];    // 16 KB
  __shared__ __align__(16) US q_s[16 * 1024];    // 32 KB
  const int b0 = blockIdx.x * 16;
  const int t = threadIdx.x;
  { // stage me tile
    const uint4* src = (const uint4*)(me_bf + (size_t)b0 * QS);
    uint4* dst = (uint4*)me_s;
    for (int i = t; i < 16 * 512 / 8; i += 256) dst[i] = src[i];
  }
  __syncthreads();
  const int wave = t >> 5, lane = t & 31, m16 = lane & 15, half = lane >> 4;
  // q GEMM: M=16, K=512, N=1024; wave owns 8 N-tiles
  for (int nt = 0; nt < 8; ++nt) {
    const int n0 = wave * 128 + nt * 16;
    v8f c = {0.f,0.f,0.f,0.f,0.f,0.f,0.f,0.f};
#pragma unroll
    for (int k0 = 0; k0 < 512; k0 += 32) {
      Frag A, Bf;
      A.u[0] = *(const uint4*)&me_s[m16 * 512 + k0 + half * 16];
      A.u[1] = *(const uint4*)&me_s[m16 * 512 + k0 + 8 + half * 16];
      const US* bp = WqT + (size_t)(n0 + m16) * 512 + k0 + half * 16;
      Bf.u[0] = *(const uint4*)bp; Bf.u[1] = *(const uint4*)(bp + 8);
      c = wmma_bf(A, Bf, c);
    }
    const float bqv = bq[n0 + m16];
#pragma unroll
    for (int v = 0; v < 8; ++v)
      q_s[(half * 8 + v) * 1024 + n0 + m16] = f2bf(c[v] + bqv);
  }
  __syncthreads();
  // p GEMM: per head (wave==head): M=16, K=128, N=512
  const int h = wave;
  for (int nt = 0; nt < 32; ++nt) {
    const int n0 = nt * 16;
    v8f c = {0.f,0.f,0.f,0.f,0.f,0.f,0.f,0.f};
#pragma unroll
    for (int k0 = 0; k0 < 128; k0 += 32) {
      Frag A, Bf;
      A.u[0] = *(const uint4*)&q_s[m16 * 1024 + h * 128 + k0 + half * 16];
      A.u[1] = *(const uint4*)&q_s[m16 * 1024 + h * 128 + k0 + 8 + half * 16];
      const US* bp = Wk_bf + ((size_t)h * 512 + n0 + m16) * 128 + k0 + half * 16;
      Bf.u[0] = *(const uint4*)bp; Bf.u[1] = *(const uint4*)(bp + 8);
      c = wmma_bf(A, Bf, c);
    }
#pragma unroll
    for (int v = 0; v < 8; ++v)
      p_ws[((size_t)(b0 + half * 8 + v) * NH + h) * 512 + n0 + m16] = f2bf(c[v]);
  }
}

// ---------------- kA: scores + softmax + r = a@other (one batch/WG) --------
__global__ __launch_bounds__(256) void kA(const float* other, const US* p_ws,
                                          US* r_ws) {
  __shared__ __align__(16) US other_s[64 * 520];   // ~66.6 KB bf16
  __shared__ __align__(16) float s_s[64 * 16];     // scores [n][h]
  __shared__ __align__(16) US a_s[16 * 64];        // softmax A matrix [h][n]
  const int b = blockIdx.x;
  const int t = threadIdx.x;
  for (int i = t; i < 64 * 16; i += 256) s_s[i] = 0.f;
  for (int i = t; i < 16 * 64; i += 256) a_s[i] = 0;
  { // stage other[b] as bf16
    const float4* src = (const float4*)(other + (size_t)b * NN * KS);
    const int row = t >> 2, cb = (t & 3) * 128;
#pragma unroll
    for (int i = 0; i < 32; ++i) {
      float4 f = src[(row * 512 + cb) / 4 + i];
      const int cix = cb + i * 4;
      other_s[row * 520 + cix + 0] = f2bf(f.x);
      other_s[row * 520 + cix + 1] = f2bf(f.y);
      other_s[row * 520 + cix + 2] = f2bf(f.z);
      other_s[row * 520 + cix + 3] = f2bf(f.w);
    }
  }
  __syncthreads();
  const int wave = t >> 5, lane = t & 31, m16 = lane & 15, half = lane >> 4;
  { // s-phase: M=64 (4 tiles) x N=16(heads,pad) ; K split across wave pairs
    const int mt = wave & 3, ks0 = (wave >> 2) * 8;
    v8f c = {0.f,0.f,0.f,0.f,0.f,0.f,0.f,0.f};
    const US* pb = p_ws + ((size_t)b * NH + m16) * 512;  // BT row = head
    const bool hv = (m16 < NH);
#pragma unroll
    for (int kk = 0; kk < 8; ++kk) {
      const int k0 = (ks0 + kk) * 32;
      Frag A, Bf;
      A.u[0] = *(const uint4*)&other_s[(mt * 16 + m16) * 520 + k0 + half * 16];
      A.u[1] = *(const uint4*)&other_s[(mt * 16 + m16) * 520 + k0 + 8 + half * 16];
      if (hv) { Bf.u[0] = *(const uint4*)(pb + k0 + half * 16);
                Bf.u[1] = *(const uint4*)(pb + k0 + half * 16 + 8); }
      else    { Bf.u[0] = make_uint4(0,0,0,0); Bf.u[1] = make_uint4(0,0,0,0); }
      c = wmma_bf(A, Bf, c);
    }
#pragma unroll
    for (int v = 0; v < 8; ++v)
      atomicAdd(&s_s[(mt * 16 + half * 8 + v) * 16 + m16], c[v]);
  }
  __syncthreads();
  { // softmax: wave == head, 64 logits across 32 lanes x2
    const float scale = 0.088388347648318447f;  // 1/sqrt(128)
    float x0 = s_s[lane * 16 + wave] * scale;
    float x1 = s_s[(lane + 32) * 16 + wave] * scale;
    float mx = fmaxf(x0, x1);
    for (int off = 16; off > 0; off >>= 1) mx = fmaxf(mx, __shfl_xor(mx, off, 32));
    float e0 = __expf(x0 - mx), e1 = __expf(x1 - mx);
    float s = e0 + e1;
    for (int off = 16; off > 0; off >>= 1) s += __shfl_xor(s, off, 32);
    const float inv = 1.f / s;
    a_s[wave * 64 + lane] = f2bf(e0 * inv);
    a_s[wave * 64 + lane + 32] = f2bf(e1 * inv);
  }
  __syncthreads();
  // r-phase: R[h][c] = a @ other : M=16(pad), K=64, N=512 (4 N-tiles/wave)
  for (int i = 0; i < 4; ++i) {
    const int n0 = (wave * 4 + i) * 16;
    v8f c = {0.f,0.f,0.f,0.f,0.f,0.f,0.f,0.f};
#pragma unroll
    for (int k0 = 0; k0 < 64; k0 += 32) {
      Frag A, Bf;
      A.u[0] = *(const uint4*)&a_s[m16 * 64 + k0 + half * 16];
      A.u[1] = *(const uint4*)&a_s[m16 * 64 + k0 + 8 + half * 16];
      US* bs = reinterpret_cast<US*>(&Bf);
#pragma unroll
      for (int j = 0; j < 16; ++j)
        bs[j] = other_s[(k0 + half * 16 + j) * 520 + n0 + m16];
      c = wmma_bf(A, Bf, c);
    }
    if (half == 0) {  // rows 0..7 are the real heads
#pragma unroll
      for (int v = 0; v < 8; ++v)
        r_ws[((size_t)b * NH + v) * 512 + n0 + m16] = f2bf(c[v]);
    }
  }
}

// ---------------- kB: out = r@Wv ; LN+ReLU ; y = cn@Wo ---------------------
__global__ __launch_bounds__(256) void kB(const US* r_ws, const US* WvT,
                                          const float* gamma, const float* beta,
                                          const US* WoT, float* out) {
  __shared__ __align__(16) float c_s[16 * 1032];   // ~66 KB, bf16 view reused
  const int b0 = blockIdx.x * 16;
  const int t = threadIdx.x;
  const int wave = t >> 5, lane = t & 31, m16 = lane & 15, half = lane >> 4;
  // out-phase: wave == head; A strip [16x512] held in 16 frags (128 VGPRs)
  const int h = wave;
  Frag A[16];
#pragma unroll
  for (int kt = 0; kt < 16; ++kt) {
    const US* ap = r_ws + ((size_t)(b0 + m16) * NH + h) * 512 + kt * 32 + half * 16;
    A[kt].u[0] = *(const uint4*)ap;
    A[kt].u[1] = *(const uint4*)(ap + 8);
  }
  for (int nt = 0; nt < 8; ++nt) {
    const int n0 = nt * 16;
    v8f c = {0.f,0.f,0.f,0.f,0.f,0.f,0.f,0.f};
#pragma unroll
    for (int kt = 0; kt < 16; ++kt) {
      Frag Bf;
      const US* bp = WvT + ((size_t)h * 128 + n0 + m16) * 512 + kt * 32 + half * 16;
      Bf.u[0] = *(const uint4*)bp; Bf.u[1] = *(const uint4*)(bp + 8);
      c = wmma_bf(A[kt], Bf, c);
    }
#pragma unroll
    for (int v = 0; v < 8; ++v)
      c_s[(half * 8 + v) * 1032 + h * 128 + n0 + m16] = c[v];
  }
  __syncthreads();
  // LayerNorm + ReLU, write bf16 in place over row start (same wave owns rows)
  for (int rr = 0; rr < 2; ++rr) {
    const int row = wave * 2 + rr;
    float vals[32]; float sum = 0.f;
#pragma unroll
    for (int j = 0; j < 32; ++j) { vals[j] = c_s[row * 1032 + j * 32 + lane]; sum += vals[j]; }
    for (int off = 16; off > 0; off >>= 1) sum += __shfl_xor(sum, off, 32);
    const float mu = sum * (1.f / 1024.f);
    float var = 0.f;
#pragma unroll
    for (int j = 0; j < 32; ++j) { float d = vals[j] - mu; var += d * d; }
    for (int off = 16; off > 0; off >>= 1) var += __shfl_xor(var, off, 32);
    const float rstd = rsqrtf(var * (1.f / 1024.f) + 1e-5f);
    US* bf = (US*)&c_s[row * 1032];
#pragma unroll
    for (int j = 0; j < 32; ++j) {
      const int col = j * 32 + lane;
      float y = (vals[j] - mu) * rstd * gamma[col] + beta[col];
      bf[col] = f2bf(fmaxf(y, 0.f));
    }
  }
  __syncthreads();
  // final GEMM: [16x1024]@WoT -> [16x512]; 4 N-tiles per wave
  const int SROW = 1032 * 2;  // row stride of bf16 view, in ushorts
  for (int i = 0; i < 4; ++i) {
    const int n0 = (wave * 4 + i) * 16;
    v8f c = {0.f,0.f,0.f,0.f,0.f,0.f,0.f,0.f};
#pragma unroll
    for (int k0 = 0; k0 < 1024; k0 += 32) {
      Frag Af, Bf;
      const US* ap = (const US*)c_s + m16 * SROW + k0 + half * 16;
      Af.u[0] = *(const uint4*)ap; Af.u[1] = *(const uint4*)(ap + 8);
      const US* bp = WoT + (size_t)(n0 + m16) * 1024 + k0 + half * 16;
      Bf.u[0] = *(const uint4*)bp; Bf.u[1] = *(const uint4*)(bp + 8);
      c = wmma_bf(Af, Bf, c);
    }
#pragma unroll
    for (int v = 0; v < 8; ++v)
      out[(size_t)(b0 + half * 8 + v) * OS + n0 + m16] = c[v];
  }
}

extern "C" void kernel_launch(void* const* d_in, const int* in_sizes, int n_in,
                              void* d_out, int out_size, void* d_ws, size_t ws_size,
                              hipStream_t stream) {
  const float* me    = (const float*)d_in[0];
  const float* other = (const float*)d_in[1];
  const float* Wq    = (const float*)d_in[2];
  const float* bq    = (const float*)d_in[3];
  const float* Wk    = (const float*)d_in[4];
  const float* Wv    = (const float*)d_in[5];
  const float* gamma = (const float*)d_in[6];
  const float* beta  = (const float*)d_in[7];
  const float* Wo    = (const float*)d_in[8];

  char* ws = (char*)d_ws;
  US* me_bf = (US*)ws;  ws += (size_t)BB * QS * 2;           // 2 MB
  US* WqT   = (US*)ws;  ws += (size_t)CC * QS * 2;           // 1 MB
  US* Wk_bf = (US*)ws;  ws += (size_t)NH * KS * HD * 2;      // 1 MB
  US* WvT   = (US*)ws;  ws += (size_t)NH * HD * KS * 2;      // 1 MB
  US* WoT   = (US*)ws;  ws += (size_t)OS * CC * 2;           // 1 MB
  US* p_ws  = (US*)ws;  ws += (size_t)BB * NH * 512 * 2;     // 16 MB
  US* r_ws  = (US*)ws;  ws += (size_t)BB * NH * 512 * 2;     // 16 MB

  const int prep_elems = BB * QS + 4 * (CC * QS);            // 3,145,728
  kPrep<<<(prep_elems + 255) / 256, 256, 0, stream>>>(me, Wq, Wk, Wv, Wo,
                                                      me_bf, WqT, Wk_bf, WvT, WoT);
  kQ<<<BB / 16, 256, 0, stream>>>(me_bf, WqT, Wk_bf, bq, p_ws);
  kA<<<BB, 256, 0, stream>>>(other, p_ws, r_ws);
  kB<<<BB / 16, 256, 0, stream>>>(r_ws, WvT, gamma, beta, WoT, (float*)d_out);
}
